// IEM_18502719111223
// MI455X (gfx1250) — matrix operand order repaired and checked by reading, not compiled, not run on hardware
//
#include <hip/hip_runtime.h>

// ---------------------------------------------------------------------------
// Problem constants (match reference setup_inputs)
// ---------------------------------------------------------------------------
#define B_     8
#define C_     128           // input channels
#define H_     192
#define W_     192
#define HW_    (H_ * W_)     // 36864 tokens
#define C3_    384           // 3*C2
#define C2_    128
#define HEADS_ 4
#define CH_    32            // channels per head
#define NTOK   128           // token tile for gram kernel (HW_ % NTOK == 0)

// ---------------------------------------------------------------------------
// WMMA / TDM vector types (CDNA5 / gfx1250, wave32)
// ---------------------------------------------------------------------------
typedef __attribute__((ext_vector_type(8)))  float        v8f;
typedef __attribute__((ext_vector_type(2)))  float        v2f;
typedef __attribute__((ext_vector_type(16))) __bf16       v16bf;
typedef __attribute__((ext_vector_type(4)))  unsigned int v4ui;
typedef __attribute__((ext_vector_type(8)))  int          v8i;
typedef __attribute__((ext_vector_type(4)))  int          v4i;

union BF16Frag { v16bf v; unsigned short u[16]; };

__device__ __forceinline__ unsigned short f2bf(float f) {
  union { float f; unsigned int u; } c; c.f = f;
  unsigned int r = c.u + 0x7FFFu + ((c.u >> 16) & 1u);  // round-to-nearest-even
  return (unsigned short)(r >> 16);
}

// bf16 A-fragment element order within a 32-wide K group (ISA 7.12.2):
// lanes<16 use kbase=0 (elements k=0..7,16..23), lanes>=16 kbase=8
// (k=8..15,24..31).  Store each m-row's 32 bf16 permuted so every half-wave's
// 16 elements are contiguous:  pos = (k&7) + ((k&8)<<1) + ((k&16)>>1)
__device__ __forceinline__ int afrag_pos(int kk) {
  return (kk & 7) + ((kk & 8) << 1) + ((kk & 16) >> 1);
}

// ---------------------------------------------------------------------------
// TDM: DMA a rows x cols fp32 tile (row stride in elements) from global to LDS.
// D# per CDNA5 ISA 8.3/8.4: group0 = {count=1, lds_addr, global_addr, type=2};
// group1 = {data_size=4B, pad ctl, tensor dims == tile dims (never OOB), dim0
// stride}. groups 2/3/4 zero (2-D tensor).  Tracked by TENSORcnt.
// pad_code pads 1 DWORD after every (2<<pad_code) DWORDs (bank-conflict pad).
// This toolchain exposes the 6-arg builtin: (g0, g1, g2, g3, g4, cpol).
// ---------------------------------------------------------------------------
__device__ __forceinline__ void tdm_load_2d(const void* gsrc, unsigned lds_off,
                                            unsigned rows, unsigned cols,
                                            unsigned row_stride,
                                            int pad_en, int pad_code) {
  unsigned long long ga = (unsigned long long)(size_t)gsrc;
  v4ui g0;
  g0.x = 1u;                                               // count=1, user D#
  g0.y = lds_off;                                          // LDS byte address
  g0.z = (unsigned)ga;                                     // global_addr[31:0]
  g0.w = (unsigned)((ga >> 32) & 0x01FFFFFFu) | (2u << 30); // [56:32] | type=2
  v8i g1;
  g1[0] = (int)((2u << 16) |                               // data_size = 4B
                ((unsigned)pad_en << 20) |                 // pad_enable
                ((unsigned)pad_code << 22));               // pad_interval (amt=1dw)
  g1[1] = (int)((cols & 0xFFFFu) << 16);                   // tensor_dim0[15:0]
  g1[2] = (int)((cols >> 16) | ((rows & 0xFFFFu) << 16));  // td0[31:16], td1[15:0]
  g1[3] = (int)((rows >> 16) | ((cols & 0xFFFFu) << 16));  // td1[31:16], tile_dim0
  g1[4] = (int)(rows & 0xFFFFu);                           // tile_dim1 (tile_dim2=0)
  g1[5] = (int)row_stride;                                 // tensor_dim0_stride[31:0]
  g1[6] = 0;
  g1[7] = 0;
  v4i z4 = {0, 0, 0, 0};
  v8i z8 = {0, 0, 0, 0, 0, 0, 0, 0};
  __builtin_amdgcn_tensor_load_to_lds(g0, g1, z4, z4, z8, 0);
}

// ---------------------------------------------------------------------------
// K0: zero accumulators (gram + norms) — harness does not re-zero ws
// ---------------------------------------------------------------------------
__global__ void k0_zero(float* __restrict__ p, int n) {
  int i = blockIdx.x * blockDim.x + threadIdx.x;
  if (i < n) p[i] = 0.f;
}

// ---------------------------------------------------------------------------
// K1: q1[b,o,n] = sum_c w1[o,c] * x[b,c,n]   (1x1 conv as GEMM, bf16 WMMA)
// Block tile 128(M) x 64(N), full K=128 resident in LDS:
//   - w1 slice converted once to bf16, stored in A-fragment order
//   - x tile (128K x 64N fp32) DMA'd by the Tensor Data Mover, then converted
//     once (block-wide) to bf16 in B-fragment order
// Every fragment load is then a single 32B v16bf load (2x ds_load_b128).
// 8 waves, each owns a 32x32 sub-tile (2x2 accumulators), 4 K-steps of 32.
// ---------------------------------------------------------------------------
__global__ __launch_bounds__(256) void k1_conv1x1(const float* __restrict__ x,
                                                  const float* __restrict__ w1,
                                                  float* __restrict__ q1) {
  __shared__ unsigned short wt[4 * 128 * 32];  // bf16 w1, [kgroup][m][fragpos]
  __shared__ float          xt[128 * 64];      // fp32 x tile [k][n] (TDM dest)
  __shared__ unsigned short xbf[4 * 64 * 32];  // bf16 x, [kgroup][n][fragpos]
  int nt0 = blockIdx.x * 64;
  int Mg  = blockIdx.y;                        // 0..2 (384 output rows / 128)
  int b   = blockIdx.z;
  const float* xb = x + (size_t)b * C_ * HW_;
  int tid = threadIdx.x;

  // stage weights: coalesced global read, bf16 convert, fragment-order store
  for (int i = tid; i < 128 * 128; i += 256) {
    int m = i >> 7, k = i & 127;
    int g = k >> 5, kk = k & 31;
    wt[(g * 128 + m) * 32 + afrag_pos(kk)] = f2bf(w1[(Mg * 128 + m) * C_ + k]);
  }
  // DMA activation tile via Tensor Data Mover (one wave issues, all consume)
  if ((tid >> 5) == 0) {
    tdm_load_2d(xb + nt0, (unsigned)(size_t)xt, /*rows*/128, /*cols*/64,
                /*stride*/HW_, /*pad*/0, 0);
    __builtin_amdgcn_s_wait_tensorcnt(0);
  }
  __syncthreads();

  // one-time bf16 conversion of the x tile into B-fragment order
  // (same permuted fragment order as A: pos = afrag_pos(k&31))
  for (int i = tid; i < 128 * 64; i += 256) {
    int k = i >> 6, n = i & 63;
    xbf[((k >> 5) * 64 + n) * 32 + afrag_pos(k & 31)] = f2bf(xt[k * 64 + n]);
  }
  __syncthreads();

  int lane = tid & 31, wv = tid >> 5;
  int mb0 = (wv >> 1) * 32;                    // wave M origin (0/32/64/96)
  int nb0 = (wv & 1) * 32;                     // wave N origin (0/32)
  int l15  = lane & 15;
  int half = (lane >> 4) << 4;                 // 16-short offset per half-wave

  v8f acc[2][2] = {};
  for (int k0 = 0; k0 < C_; k0 += 32) {
    int g = k0 >> 5;
    BF16Frag a[2], bb[2];
#pragma unroll
    for (int i = 0; i < 2; ++i) {
      int mloc = mb0 + i * 16 + l15;
      a[i].v = *reinterpret_cast<const v16bf*>(&wt[(g * 128 + mloc) * 32 + half]);
    }
#pragma unroll
    for (int jj = 0; jj < 2; ++jj) {
      int nloc = nb0 + jj * 16 + l15;
      bb[jj].v = *reinterpret_cast<const v16bf*>(&xbf[(g * 64 + nloc) * 32 + half]);
    }
#pragma unroll
    for (int i = 0; i < 2; ++i)
#pragma unroll
      for (int jj = 0; jj < 2; ++jj)
        acc[i][jj] = __builtin_amdgcn_wmma_f32_16x16x32_bf16(
            false, a[i].v, false, bb[jj].v, (short)0, acc[i][jj], false, false);
  }

  float* qb = q1 + (size_t)b * C3_ * HW_;
  int mlane = (lane >> 4) * 8;
#pragma unroll
  for (int i = 0; i < 2; ++i)
#pragma unroll
    for (int jj = 0; jj < 2; ++jj) {
      int ncol = nt0 + nb0 + jj * 16 + l15;
#pragma unroll
      for (int rr = 0; rr < 8; ++rr) {
        int orow = Mg * 128 + mb0 + i * 16 + mlane + rr;
        qb[(size_t)orow * HW_ + ncol] = acc[i][jj][rr];
      }
    }
}

// ---------------------------------------------------------------------------
// K2: depthwise 3x3 SAME (cross-correlation, zero pad), one thread per output
// ---------------------------------------------------------------------------
__global__ __launch_bounds__(256) void k2_dw(const float* __restrict__ q1,
                                             const float* __restrict__ wd,
                                             float* __restrict__ q1d) {
  size_t idx = (size_t)blockIdx.x * 256 + threadIdx.x;
  const size_t total = (size_t)B_ * C3_ * HW_;
  if (idx >= total) return;
  int n  = (int)(idx % HW_);
  int oc = (int)((idx / HW_) % C3_);
  int h = n / W_, w = n % W_;
  const float* src = q1 + (idx - n);
  const float* wk  = wd + oc * 9;
  float s = 0.f;
#pragma unroll
  for (int dy = -1; dy <= 1; ++dy) {
    int hh = h + dy;
    if (hh < 0 || hh >= H_) continue;
#pragma unroll
    for (int dx = -1; dx <= 1; ++dx) {
      int ww = w + dx;
      if (ww < 0 || ww >= W_) continue;
      s += wk[(dy + 1) * 3 + (dx + 1)] * src[hh * W_ + ww];
    }
  }
  q1d[idx] = s;
}

// ---------------------------------------------------------------------------
// K3: per (b,head) Gram matrix gram = q_dw @ k_dw^T (32x32, K = HW tokens)
// fp32 WMMA 16x16x4. Tiles DMA'd by TDM with pad_interval=128dw -> LDS row
// stride 129 floats -> conflict-free column (fragment) reads.
// 8 waves x 16 tokens; partial 32x32 results accumulated by global atomicAdd.
// ---------------------------------------------------------------------------
__global__ __launch_bounds__(256) void k3_gram(const float* __restrict__ q1d,
                                               float* __restrict__ gram) {
  __shared__ float qt[CH_ * (NTOK + 1)];
  __shared__ float kt[CH_ * (NTOK + 1)];
  int bh = blockIdx.y;                 // b*HEADS_+h
  int b = bh / HEADS_, h = bh % HEADS_;
  int n0 = blockIdx.x * NTOK;
  const float* qbase = q1d + ((size_t)b * C3_ + h * CH_) * HW_ + n0;
  const float* kbase = q1d + ((size_t)b * C3_ + C2_ + h * CH_) * HW_ + n0;
  int tid = threadIdx.x;
  if ((tid >> 5) == 0) {
    tdm_load_2d(qbase, (unsigned)(size_t)qt, CH_, NTOK, HW_, 1, 6); // pad@128dw
    tdm_load_2d(kbase, (unsigned)(size_t)kt, CH_, NTOK, HW_, 1, 6);
    __builtin_amdgcn_s_wait_tensorcnt(0);
  }
  __syncthreads();

  int wv = tid >> 5, lane = tid & 31;
  int col = lane & 15;
  int kb  = (lane >> 4) * 2;           // f32 A/B: lanes 16-31 carry K+2
  int nb  = wv * (NTOK / 8);           // 16 tokens per wave
  v8f acc[2][2] = {};
  for (int kk = 0; kk < NTOK / 8; kk += 4) {
    int nn = nb + kk + kb;
    v2f a0, a1, b0, b1;
    a0.x = qt[col * (NTOK + 1) + nn];        a0.y = qt[col * (NTOK + 1) + nn + 1];
    a1.x = qt[(16 + col) * (NTOK + 1) + nn]; a1.y = qt[(16 + col) * (NTOK + 1) + nn + 1];
    b0.x = kt[col * (NTOK + 1) + nn];        b0.y = kt[col * (NTOK + 1) + nn + 1];
    b1.x = kt[(16 + col) * (NTOK + 1) + nn]; b1.y = kt[(16 + col) * (NTOK + 1) + nn + 1];
    acc[0][0] = __builtin_amdgcn_wmma_f32_16x16x4_f32(false, a0, false, b0, (short)0, acc[0][0], false, false);
    acc[0][1] = __builtin_amdgcn_wmma_f32_16x16x4_f32(false, a0, false, b1, (short)0, acc[0][1], false, false);
    acc[1][0] = __builtin_amdgcn_wmma_f32_16x16x4_f32(false, a1, false, b0, (short)0, acc[1][0], false, false);
    acc[1][1] = __builtin_amdgcn_wmma_f32_16x16x4_f32(false, a1, false, b1, (short)0, acc[1][1], false, false);
  }
  float* g = gram + (size_t)bh * CH_ * CH_;
  int mb = (lane >> 4) * 8;
#pragma unroll
  for (int i = 0; i < 2; ++i)
#pragma unroll
    for (int j = 0; j < 2; ++j)
#pragma unroll
      for (int rr = 0; rr < 8; ++rr)
        atomicAdd(&g[(i * 16 + mb + rr) * CH_ + j * 16 + col], acc[i][j][rr]);
}

// ---------------------------------------------------------------------------
// K3b: squared L2 norms of q,k channels over tokens; one block per channel
// ---------------------------------------------------------------------------
__global__ __launch_bounds__(256) void k3b_norm(const float* __restrict__ q1d,
                                                float* __restrict__ qn2,
                                                float* __restrict__ kn2) {
  int id = blockIdx.x;                    // [0, B_*2*C2_)
  int ch = id % C2_;
  int qk = (id / C2_) & 1;
  int b  = id / (2 * C2_);
  const float* p = q1d + ((size_t)b * C3_ + qk * C2_ + ch) * HW_;
  float s = 0.f;
  for (int n = threadIdx.x; n < HW_; n += 256) { float v = p[n]; s += v * v; }
  __shared__ float red[256];
  red[threadIdx.x] = s; __syncthreads();
  for (int off = 128; off > 0; off >>= 1) {
    if (threadIdx.x < off) red[threadIdx.x] += red[threadIdx.x + off];
    __syncthreads();
  }
  if (threadIdx.x == 0) (qk ? kn2 : qn2)[b * C2_ + ch] = red[0];
}

// ---------------------------------------------------------------------------
// K4: normalize gram by ||q_c||*||k_d|| (eps-clamped), scale by a1, softmax
// ---------------------------------------------------------------------------
__global__ void k4_softmax(const float* __restrict__ gram,
                           const float* __restrict__ qn2,
                           const float* __restrict__ kn2,
                           const float* __restrict__ a1,
                           float* __restrict__ attn) {
  int bh = blockIdx.x; int b = bh / HEADS_, h = bh % HEADS_;
  int cRow = threadIdx.x;
  const float* g = gram + (size_t)bh * CH_ * CH_;
  float qn = fmaxf(sqrtf(qn2[b * C2_ + h * CH_ + cRow]), 1e-12f);
  float temp = a1[h];
  float row[CH_];
  float mx = -1e30f;
  for (int d = 0; d < CH_; ++d) {
    float kn = fmaxf(sqrtf(kn2[b * C2_ + h * CH_ + d]), 1e-12f);
    float v = g[cRow * CH_ + d] / (qn * kn) * temp;
    row[d] = v; mx = fmaxf(mx, v);
  }
  float sum = 0.f;
  for (int d = 0; d < CH_; ++d) { row[d] = __expf(row[d] - mx); sum += row[d]; }
  float inv = 1.f / sum;
  float* o = attn + (size_t)bh * CH_ * CH_;
  for (int d = 0; d < CH_; ++d) o[cRow * CH_ + d] = row[d] * inv;
}

// ---------------------------------------------------------------------------
// K4b: fold projection through attention:  M[b] = wproj @ blockdiag(attn[b,h])
// M[o][h*32+d] = sum_r wproj[o, h*32+r] * attn[b,h,r,d]
// ---------------------------------------------------------------------------
__global__ __launch_bounds__(128) void k4b_mmat(const float* __restrict__ wproj,
                                                const float* __restrict__ attn,
                                                float* __restrict__ Mmat) {
  int b = blockIdx.x;
  __shared__ float at[HEADS_][CH_][CH_];
  int tid = threadIdx.x;
  for (int i = tid; i < HEADS_ * CH_ * CH_; i += 128)
    (&at[0][0][0])[i] = attn[(size_t)b * HEADS_ * CH_ * CH_ + i];
  __syncthreads();
  int o = tid;
  float* Mrow = Mmat + ((size_t)b * C_ + o) * C2_;
  for (int j = 0; j < C2_; ++j) {
    int hh = j / CH_, jj = j % CH_;
    float s = 0.f;
#pragma unroll
    for (int i = 0; i < CH_; ++i)
      s += wproj[o * C2_ + hh * CH_ + i] * at[hh][i][jj];
    Mrow[j] = s;
  }
}

// ---------------------------------------------------------------------------
// K5: y = x + M[b] @ v_dw  (attn@v + project_out + residual, bf16 WMMA)
// Same block structure as K1: M[b] staged bf16 in A-fragment order, v tile
// DMA'd by TDM then converted once to bf16 B-fragment order.
// ---------------------------------------------------------------------------
__global__ __launch_bounds__(256) void k5_out(const float* __restrict__ x,
                                              const float* __restrict__ Mmat,
                                              const float* __restrict__ q1d,
                                              float* __restrict__ y) {
  __shared__ unsigned short mt[4 * 128 * 32]; // bf16 M[b], [kgroup][o][fragpos]
  __shared__ float          vt[128 * 64];     // fp32 v tile [k][n] (TDM dest)
  __shared__ unsigned short vbf[4 * 64 * 32]; // bf16 v, [kgroup][n][fragpos]
  int nt0 = blockIdx.x * 64;
  int b   = blockIdx.y;
  const float* vb = q1d + ((size_t)b * C3_ + 2 * C2_) * HW_;
  const float* Mb = Mmat + (size_t)b * C_ * C2_;
  int tid = threadIdx.x;

  for (int i = tid; i < 128 * 128; i += 256) {
    int m = i >> 7, k = i & 127;
    int g = k >> 5, kk = k & 31;
    mt[(g * 128 + m) * 32 + afrag_pos(kk)] = f2bf(Mb[m * C2_ + k]);
  }
  if ((tid >> 5) == 0) {
    tdm_load_2d(vb + nt0, (unsigned)(size_t)vt, 128, 64, HW_, 0, 0);
    __builtin_amdgcn_s_wait_tensorcnt(0);
  }
  __syncthreads();

  for (int i = tid; i < 128 * 64; i += 256) {
    int k = i >> 6, n = i & 63;
    vbf[((k >> 5) * 64 + n) * 32 + afrag_pos(k & 31)] = f2bf(vt[k * 64 + n]);
  }
  __syncthreads();

  int lane = tid & 31, wv = tid >> 5;
  int mb0 = (wv >> 1) * 32;
  int nb0 = (wv & 1) * 32;
  int l15  = lane & 15;
  int half = (lane >> 4) << 4;

  v8f acc[2][2] = {};
  for (int k0 = 0; k0 < C2_; k0 += 32) {
    int g = k0 >> 5;
    BF16Frag a[2], bb[2];
#pragma unroll
    for (int i = 0; i < 2; ++i) {
      int mloc = mb0 + i * 16 + l15;
      a[i].v = *reinterpret_cast<const v16bf*>(&mt[(g * 128 + mloc) * 32 + half]);
    }
#pragma unroll
    for (int jj = 0; jj < 2; ++jj) {
      int nloc = nb0 + jj * 16 + l15;
      bb[jj].v = *reinterpret_cast<const v16bf*>(&vbf[(g * 64 + nloc) * 32 + half]);
    }
#pragma unroll
    for (int i = 0; i < 2; ++i)
#pragma unroll
      for (int jj = 0; jj < 2; ++jj)
        acc[i][jj] = __builtin_amdgcn_wmma_f32_16x16x32_bf16(
            false, a[i].v, false, bb[jj].v, (short)0, acc[i][jj], false, false);
  }

  const float* xb = x + (size_t)b * C_ * HW_;
  float* yb = y + (size_t)b * C_ * HW_;
  int mlane = (lane >> 4) * 8;
#pragma unroll
  for (int i = 0; i < 2; ++i)
#pragma unroll
    for (int jj = 0; jj < 2; ++jj) {
      int ncol = nt0 + nb0 + jj * 16 + l15;
#pragma unroll
      for (int rr = 0; rr < 8; ++rr) {
        size_t oo = (size_t)(mb0 + i * 16 + mlane + rr) * HW_ + ncol;
        yb[oo] = xb[oo] + acc[i][jj][rr];
      }
    }
}

// ---------------------------------------------------------------------------
// Launch.  ws layout (floats): q1 | q1d | gram | qn2 | kn2 | attn | M
// ---------------------------------------------------------------------------
extern "C" void kernel_launch(void* const* d_in, const int* in_sizes, int n_in,
                              void* d_out, int out_size, void* d_ws, size_t ws_size,
                              hipStream_t stream) {
  const float* x     = (const float*)d_in[0];
  const float* w1    = (const float*)d_in[1];
  const float* wd    = (const float*)d_in[2];
  const float* wproj = (const float*)d_in[3];
  const float* a1    = (const float*)d_in[4];
  float* out = (float*)d_out;
  float* ws  = (float*)d_ws;

  size_t off = 0;
  float* q1   = ws + off; off += (size_t)B_ * C3_ * HW_;
  float* q1d  = ws + off; off += (size_t)B_ * C3_ * HW_;
  float* gram = ws + off; off += (size_t)B_ * HEADS_ * CH_ * CH_;
  float* qn2  = ws + off; off += (size_t)B_ * C2_;
  float* kn2  = ws + off; off += (size_t)B_ * C2_;
  float* attn = ws + off; off += (size_t)B_ * HEADS_ * CH_ * CH_;
  float* Mmat = ws + off; off += (size_t)B_ * C_ * C2_;

  int zn = B_ * HEADS_ * CH_ * CH_ + 2 * B_ * C2_;
  k0_zero<<<(zn + 255) / 256, 256, 0, stream>>>(gram, zn);

  dim3 g1(HW_ / 64, C3_ / 128, B_);
  k1_conv1x1<<<g1, 256, 0, stream>>>(x, w1, q1);

  size_t tot = (size_t)B_ * C3_ * HW_;
  k2_dw<<<(unsigned)((tot + 255) / 256), 256, 0, stream>>>(q1, wd, q1d);

  dim3 g3(HW_ / NTOK, B_ * HEADS_);
  k3_gram<<<g3, 256, 0, stream>>>(q1d, gram);
  k3b_norm<<<B_ * 2 * C2_, 256, 0, stream>>>(q1d, qn2, kn2);

  k4_softmax<<<B_ * HEADS_, 32, 0, stream>>>(gram, qn2, kn2, a1, attn);
  k4b_mmat<<<B_, 128, 0, stream>>>(wproj, attn, Mmat);

  dim3 g5(HW_ / 64, B_);
  k5_out<<<g5, 256, 0, stream>>>(x, Mmat, q1d, out);
}
